// FuturePoseRNN_85864986181752
// MI455X (gfx1250) — compile-verified, matrix operand order checked by reading, not run on hardware
//
#include <hip/hip_runtime.h>

#define POSE 13
#define HID 256
#define T_STEPS 48
#define MTILE 32
#define NTHREADS 128
#define B_TOTAL 4096

typedef __attribute__((ext_vector_type(16))) _Float16 v16h;
typedef __attribute__((ext_vector_type(8)))  _Float16 v8h;
typedef __attribute__((ext_vector_type(8)))  float    v8f;

__device__ __forceinline__ v8f wmma16(v16h a, v16h b, v8f c) {
  // D = A(16x32 f16) * B(32x16 f16) + C(16x16 f32)
  return __builtin_amdgcn_wmma_f32_16x16x32_f16(false, a, false, b, (short)0, c,
                                                false, false);
}

// A-fragment from a row-major f16 matrix in LDS (row stride `stride` halves).
// CDNA5 16-bit operand layout: lane L in [0,16) holds row L, K = k0+{0..7},
// k0+{16..23}; lane L+16 holds row L with K offset +8.
__device__ __forceinline__ v16h load_fragA(const _Float16* p, int row0, int k0,
                                           int stride, int lane) {
  int r  = row0 + (lane & 15);
  int kb = k0 + ((lane >> 4) << 3);
  const _Float16* q = p + r * stride + kb;
  v8h lo = *(const v8h*)(q);
  v8h hi = *(const v8h*)(q + 16);
  return __builtin_shufflevector(lo, hi, 0,1,2,3,4,5,6,7,8,9,10,11,12,13,14,15);
}

// B-fragment from pre-swizzled (fragment-major) weights: each 16(N)x32(K)
// tile is a contiguous 512-half block holding the exact 32-lane x 16-half
// register image -> one fully-coalesced 1KB wave burst per fragment.
__device__ __forceinline__ v16h load_fragB(const _Float16* base, int block,
                                           int lane) {
  return *(const v16h*)(base + block * 512 + lane * 16);
}

__device__ __forceinline__ v8f splat8(float v) {
  v8f r;
  #pragma unroll
  for (int i = 0; i < 8; ++i) r[i] = v;
  return r;
}

// Map a flat index of a swizzled array to (row, col) of the source matrix.
__device__ __forceinline__ void swz_coord(int idx, int ktiles, int& block,
                                          int& row_in_tile, int& col_in_tile) {
  block = idx >> 9;
  int lane = (idx >> 4) & 31;
  int i = idx & 15;
  row_in_tile = lane & 15;
  col_in_tile = ((lane >> 4) << 3) + (i & 7) + ((i >> 3) << 4);
  (void)ktiles;
}

// ---- one-time weight conversion fp32 -> swizzled f16 (+ bias fusion) ----
__global__ void prep_weights(const float* __restrict__ W_in,
                             const float* __restrict__ W_ih, const float* __restrict__ b_ih,
                             const float* __restrict__ W_hh, const float* __restrict__ b_hh,
                             const float* __restrict__ W_out,
                             _Float16* __restrict__ Wih_h, _Float16* __restrict__ Whh_h,
                             _Float16* __restrict__ Wp_h,  _Float16* __restrict__ Wc_h,
                             _Float16* __restrict__ Wout_h,
                             float* __restrict__ bias_r, float* __restrict__ bias_z,
                             float* __restrict__ bias_in, float* __restrict__ bias_hn) {
  int stride = gridDim.x * blockDim.x;
  int i0 = blockIdx.x * blockDim.x + threadIdx.x;

  // W_ih / W_hh: 768x256, ktiles = 8 -> 384 blocks
  for (int idx = i0; idx < 768 * 256; idx += stride) {
    int block, rt, ct;
    swz_coord(idx, 8, block, rt, ct);
    int row = (block >> 3) * 16 + rt;
    int col = (block & 7) * 32 + ct;
    Wih_h[idx] = (_Float16)W_ih[row * 256 + col];
    Whh_h[idx] = (_Float16)W_hh[row * 256 + col];
  }
  // pose part of W_in: 256 rows x 32 K (cols >= 13 zero), ktiles = 1
  for (int idx = i0; idx < 256 * 32; idx += stride) {
    int block, rt, ct;
    swz_coord(idx, 1, block, rt, ct);
    int row = block * 16 + rt;
    Wp_h[idx] = (ct < POSE) ? (_Float16)W_in[row * 269 + ct] : (_Float16)0.f;
  }
  // context part of W_in: 256 rows x 256 K, ktiles = 8
  for (int idx = i0; idx < 256 * 256; idx += stride) {
    int block, rt, ct;
    swz_coord(idx, 8, block, rt, ct);
    int row = (block >> 3) * 16 + rt;
    int col = (block & 7) * 32 + ct;
    Wc_h[idx] = (_Float16)W_in[row * 269 + 13 + col];
  }
  // W_out: 16 rows (>=13 zero) x 256 K, single N tile, ktiles = 8
  for (int idx = i0; idx < 8 * 512; idx += stride) {
    int block, rt, ct;
    swz_coord(idx, 8, block, rt, ct);
    int col = block * 32 + ct;
    Wout_h[idx] = (rt < POSE) ? (_Float16)W_out[rt * 256 + col] : (_Float16)0.f;
  }
  for (int i = i0; i < 256; i += stride) {
    bias_r[i]  = b_ih[i]       + b_hh[i];
    bias_z[i]  = b_ih[256 + i] + b_hh[256 + i];
    bias_in[i] = b_ih[512 + i];
    bias_hn[i] = b_hh[512 + i];
  }
}

// ---- main rollout: one WG owns a 32-row batch tile for all 48 steps ----
__global__ __launch_bounds__(NTHREADS)
void rnn_rollout(const float* __restrict__ last_pose,
                 const float* __restrict__ context,
                 const float* __restrict__ b_in,
                 const float* __restrict__ b_out,
                 const _Float16* __restrict__ Wih,
                 const _Float16* __restrict__ Whh,
                 const _Float16* __restrict__ Wp,
                 const _Float16* __restrict__ Wc,
                 const _Float16* __restrict__ Wout,
                 const float* __restrict__ bias_r,
                 const float* __restrict__ bias_z,
                 const float* __restrict__ bias_in,
                 const float* __restrict__ bias_hn,
                 float* __restrict__ out) {
  __shared__ _Float16 x16[MTILE * HID];       // 16 KB  activations x (f16)
  __shared__ _Float16 h16[2][MTILE * HID];    // 32 KB  hidden state (f16, dbl-buf)
  __shared__ float    h32[2][MTILE * HID];    // 64 KB  hidden state (f32, dbl-buf)
  __shared__ float    xcs[MTILE * HID];       // 32 KB  context @ Wc^T + b_in
  __shared__ float    pose32[MTILE * 16];     //  2 KB
  __shared__ _Float16 pose16[MTILE * 32];     //  2 KB  K-padded A-operand pose
  __shared__ float    delta[MTILE * 16];      //  2 KB

  const int tid    = threadIdx.x;
  const int lane   = tid & 31;
  const int wave   = tid >> 5;
  const int batch0 = blockIdx.x * MTILE;
  const int col    = lane & 15;           // C-layout column for this lane
  const int rowoff = (lane >> 4) << 3;    // C-layout row base for this lane

  // ---------- init: context->f16 (staged in x16), zero h, load pose ----------
  for (int i = tid; i < MTILE * HID; i += NTHREADS) {
    int r = i >> 8, c = i & 255;
    x16[i] = (_Float16)context[(long)(batch0 + r) * HID + c];
    h16[0][i] = (_Float16)0.f; h16[1][i] = (_Float16)0.f;
    h32[0][i] = 0.f;           h32[1][i] = 0.f;
  }
  for (int i = tid; i < MTILE * 16; i += NTHREADS) {
    int r = i >> 4, c = i & 15;
    pose32[i] = (c < POSE) ? last_pose[(long)(batch0 + r) * POSE + c] : 0.f;
    delta[i] = 0.f;
  }
  for (int i = tid; i < MTILE * 32; i += NTHREADS) {
    int r = i >> 5, c = i & 31;
    pose16[i] = (c < POSE) ? (_Float16)last_pose[(long)(batch0 + r) * POSE + c]
                           : (_Float16)0.f;
  }
  __syncthreads();

  // ---------- once: xc = context @ Wc^T + b_in  (K = 256) ----------
  for (int nt = wave * 4; nt < wave * 4 + 4; ++nt) {
    int n0 = nt * 16;
    for (int mt = 0; mt < 2; ++mt) {
      int m0 = mt * 16;
      v8f acc = splat8(b_in[n0 + col]);
      for (int kt = 0; kt < 8; ++kt) {
        v16h a = load_fragA(x16, m0, kt * 32, HID, lane);
        v16h b = load_fragB(Wc, nt * 8 + kt, lane);
        acc = wmma16(a, b, acc);
      }
      #pragma unroll
      for (int r = 0; r < 8; ++r)
        xcs[(m0 + rowoff + r) * HID + n0 + col] = acc[r];
    }
  }
  __syncthreads();

  int p = 0;
  for (int t = 0; t < T_STEPS; ++t) {
    // ---- Phase A: x = relu(pose @ Wp^T + xc)  (one K=32 WMMA per tile) ----
    for (int nt = wave * 4; nt < wave * 4 + 4; ++nt) {
      int n0 = nt * 16;
      for (int mt = 0; mt < 2; ++mt) {
        int m0 = mt * 16;
        v16h a = load_fragA(pose16, m0, 0, 32, lane);
        v16h b = load_fragB(Wp, nt, lane);
        v8f acc = wmma16(a, b, splat8(0.f));
        #pragma unroll
        for (int r = 0; r < 8; ++r) {
          int idx = (m0 + rowoff + r) * HID + n0 + col;
          x16[idx] = (_Float16)fmaxf(acc[r] + xcs[idx], 0.f);
        }
      }
    }
    __syncthreads();

    // ---- Phase B: fused gates; B-fragments shared across both M-tiles ----
    const _Float16* h16o = h16[p];
    const float*    h32o = h32[p];
    _Float16* h16n = h16[p ^ 1];
    float*    h32n = h32[p ^ 1];
    for (int nt = wave * 4; nt < wave * 4 + 4; ++nt) {
      int n0 = nt * 16;
      float br = bias_r[n0 + col],  bz = bias_z[n0 + col];
      float bi = bias_in[n0 + col], bh = bias_hn[n0 + col];
      v8f ar0 = splat8(br), ar1 = splat8(br);
      v8f az0 = splat8(bz), az1 = splat8(bz);
      v8f ai0 = splat8(bi), ai1 = splat8(bi);
      v8f an0 = splat8(bh), an1 = splat8(bh);
      for (int kt = 0; kt < 8; ++kt) {
        // r/z/n gate rows live at nt, nt+16, nt+32 tiles of the 768-row weights
        v16h wr_x = load_fragB(Wih, (nt +  0) * 8 + kt, lane);
        v16h wr_h = load_fragB(Whh, (nt +  0) * 8 + kt, lane);
        v16h wz_x = load_fragB(Wih, (nt + 16) * 8 + kt, lane);
        v16h wz_h = load_fragB(Whh, (nt + 16) * 8 + kt, lane);
        v16h wn_x = load_fragB(Wih, (nt + 32) * 8 + kt, lane);
        v16h wn_h = load_fragB(Whh, (nt + 32) * 8 + kt, lane);
        v16h ax0 = load_fragA(x16,  0,  kt * 32, HID, lane);
        v16h ah0 = load_fragA(h16o, 0,  kt * 32, HID, lane);
        v16h ax1 = load_fragA(x16,  16, kt * 32, HID, lane);
        v16h ah1 = load_fragA(h16o, 16, kt * 32, HID, lane);
        ar0 = wmma16(ax0, wr_x, ar0); ar0 = wmma16(ah0, wr_h, ar0);
        az0 = wmma16(ax0, wz_x, az0); az0 = wmma16(ah0, wz_h, az0);
        ai0 = wmma16(ax0, wn_x, ai0); an0 = wmma16(ah0, wn_h, an0);
        ar1 = wmma16(ax1, wr_x, ar1); ar1 = wmma16(ah1, wr_h, ar1);
        az1 = wmma16(ax1, wz_x, az1); az1 = wmma16(ah1, wz_h, az1);
        ai1 = wmma16(ax1, wn_x, ai1); an1 = wmma16(ah1, wn_h, an1);
      }
      #pragma unroll
      for (int r = 0; r < 8; ++r) {
        {
          int idx = (rowoff + r) * HID + n0 + col;
          float rr = 1.f / (1.f + __expf(-ar0[r]));
          float zz = 1.f / (1.f + __expf(-az0[r]));
          float nn = tanhf(ai0[r] + rr * an0[r]);
          float hv = (1.f - zz) * nn + zz * h32o[idx];
          h32n[idx] = hv; h16n[idx] = (_Float16)hv;
        }
        {
          int idx = (16 + rowoff + r) * HID + n0 + col;
          float rr = 1.f / (1.f + __expf(-ar1[r]));
          float zz = 1.f / (1.f + __expf(-az1[r]));
          float nn = tanhf(ai1[r] + rr * an1[r]);
          float hv = (1.f - zz) * nn + zz * h32o[idx];
          h32n[idx] = hv; h16n[idx] = (_Float16)hv;
        }
      }
    }
    __syncthreads();

    // ---- Phase C: delta = h_new @ Wout^T (N padded to 16) ----
    if (wave < 2) {
      int m0 = wave * 16;
      v8f acc = splat8(0.f);
      for (int kt = 0; kt < 8; ++kt) {
        v16h a = load_fragA(h16n, m0, kt * 32, HID, lane);
        v16h b = load_fragB(Wout, kt, lane);
        acc = wmma16(a, b, acc);
      }
      #pragma unroll
      for (int r = 0; r < 8; ++r)
        delta[(m0 + rowoff + r) * 16 + col] = acc[r];
    }
    __syncthreads();

    // ---- pose integrate + quaternion renorm + store output ----
    if (tid < MTILE) {
      float pv[POSE];
      #pragma unroll
      for (int c = 0; c < POSE; ++c)
        pv[c] = pose32[tid * 16 + c] + delta[tid * 16 + c] + b_out[c];
      float s  = pv[3]*pv[3] + pv[4]*pv[4] + pv[5]*pv[5] + pv[6]*pv[6];
      float qn = fmaxf(sqrtf(s), 1e-12f);
      #pragma unroll
      for (int c = 3; c < 7; ++c) pv[c] /= qn;
      long base = ((long)(batch0 + tid) * T_STEPS + t) * POSE;
      #pragma unroll
      for (int c = 0; c < POSE; ++c) {
        out[base + c]         = pv[c];
        pose32[tid * 16 + c]  = pv[c];
        pose16[tid * 32 + c]  = (_Float16)pv[c];
      }
    }
    __syncthreads();
    p ^= 1;
  }
}

extern "C" void kernel_launch(void* const* d_in, const int* in_sizes, int n_in,
                              void* d_out, int out_size, void* d_ws, size_t ws_size,
                              hipStream_t stream) {
  (void)in_sizes; (void)n_in; (void)out_size; (void)ws_size;
  const float* last_pose = (const float*)d_in[0];
  const float* context   = (const float*)d_in[1];
  // d_in[2] = num_future (== 48, baked into T_STEPS)
  const float* W_in  = (const float*)d_in[3];
  const float* b_in  = (const float*)d_in[4];
  const float* W_ih  = (const float*)d_in[5];
  const float* b_ih  = (const float*)d_in[6];
  const float* W_hh  = (const float*)d_in[7];
  const float* b_hh  = (const float*)d_in[8];
  const float* W_out = (const float*)d_in[9];
  const float* b_out = (const float*)d_in[10];

  char* ws = (char*)d_ws;
  _Float16* Wih_h  = (_Float16*)(ws + 0);        // 768*256 f16 (swizzled)
  _Float16* Whh_h  = (_Float16*)(ws + 393216);   // 768*256 f16 (swizzled)
  _Float16* Wp_h   = (_Float16*)(ws + 786432);   // 256*32  f16 (swizzled)
  _Float16* Wc_h   = (_Float16*)(ws + 802816);   // 256*256 f16 (swizzled)
  _Float16* Wout_h = (_Float16*)(ws + 933888);   // 16*256  f16 (swizzled)
  float* bias_r  = (float*)(ws + 942080);
  float* bias_z  = (float*)(ws + 943104);
  float* bias_in = (float*)(ws + 944128);
  float* bias_hn = (float*)(ws + 945152);

  prep_weights<<<dim3(512), dim3(256), 0, stream>>>(
      W_in, W_ih, b_ih, W_hh, b_hh, W_out,
      Wih_h, Whh_h, Wp_h, Wc_h, Wout_h, bias_r, bias_z, bias_in, bias_hn);

  rnn_rollout<<<dim3(B_TOTAL / MTILE), dim3(NTHREADS), 0, stream>>>(
      last_pose, context, b_in, b_out,
      Wih_h, Whh_h, Wp_h, Wc_h, Wout_h,
      bias_r, bias_z, bias_in, bias_hn, (float*)d_out);
}
